// VectorQuantizer_57578331570534
// MI455X (gfx1250) — matrix operand order repaired and checked by reading, not compiled, not run on hardware
//
#include <hip/hip_runtime.h>
#include <hip/hip_bf16.h>

#define N_TOK   524288
#define N_CODE  512
#define DIM     64
#define WAVES   8      // waves per block (256 threads, wave32)
#define TPW     4      // 16-token tiles per wave

typedef __attribute__((ext_vector_type(16))) __bf16 v16bf;
typedef __attribute__((ext_vector_type(8)))  float  v8f;

#define SCORE_BIAS 512.0f   // makes all packed scores positive (|s| <~ 200)

// Load one lane's slice of the 16x32 bf16 A tile (contiguous K):
// lane<16: row=lane,    K = {ka..ka+15} with ka=0
// lane>=16: row=lane-16, same with ka=16. base = ks*32 + ka.
__device__ __forceinline__ v16bf load_a_slice(const float* __restrict__ zrow, int base) {
    const float4* p = (const float4*)(zrow + base);
    const float4* q = (const float4*)(zrow + base + 8);
    float4 u0 = p[0], u1 = p[1], u2 = q[0], u3 = q[1];
    v16bf a;
    a[0] = (__bf16)u0.x; a[1] = (__bf16)u0.y; a[2]  = (__bf16)u0.z; a[3]  = (__bf16)u0.w;
    a[4] = (__bf16)u1.x; a[5] = (__bf16)u1.y; a[6]  = (__bf16)u1.z; a[7]  = (__bf16)u1.w;
    a[8] = (__bf16)u2.x; a[9] = (__bf16)u2.y; a[10] = (__bf16)u2.z; a[11] = (__bf16)u2.w;
    a[12] = (__bf16)u3.x; a[13] = (__bf16)u3.y; a[14] = (__bf16)u3.z; a[15] = (__bf16)u3.w;
    return a;
}

__global__ void __launch_bounds__(256) vq_init_kernel(float* loss_acc) {
    if (threadIdx.x == 0) *loss_acc = 0.0f;
}

__global__ void __launch_bounds__(256) vq_main_kernel(const float* __restrict__ ze,
                                                      const float* __restrict__ emb,
                                                      float* __restrict__ out,
                                                      float* __restrict__ loss_acc) {
    extern __shared__ char smem[];
    __bf16* sE = (__bf16*)smem;                 // 512 x 64 bf16 codebook (64 KB)
    float*  sN = (float*)(smem + N_CODE * DIM * sizeof(__bf16)); // 512 - ||e||^2 (2 KB)

    // Stage codebook: bf16 copy + biased negative squared norms.
    for (int r = threadIdx.x; r < N_CODE; r += 256) {
        const float4* src = (const float4*)(emb + r * DIM);
        __bf16* dst = sE + r * DIM;
        float nrm = 0.0f;
        #pragma unroll
        for (int j = 0; j < DIM / 4; ++j) {
            float4 v = src[j];
            nrm += v.x * v.x + v.y * v.y + v.z * v.z + v.w * v.w;
            dst[4 * j + 0] = (__bf16)v.x;
            dst[4 * j + 1] = (__bf16)v.y;
            dst[4 * j + 2] = (__bf16)v.z;
            dst[4 * j + 3] = (__bf16)v.w;
        }
        sN[r] = SCORE_BIAS - nrm;   // fma folds bias + (-norm) in one op
    }
    __syncthreads();

    const int lane = threadIdx.x & 31;
    const int wave = threadIdx.x >> 5;
    const int gw   = blockIdx.x * WAVES + wave;

    float* out_ix = out + (size_t)N_TOK * DIM;
    const int ka = (lane & 16) ? 16 : 0;   // A-layout K sub-offset per half-wave
    const int kb = (lane & 16) ? 16 : 0;   // B-layout K sub-offset per half-wave
    const int col = lane & 15;             // B/C column within a 16-code tile

    float lsum = 0.0f;

    for (int t = 0; t < TPW; ++t) {
        const int tok0 = (gw * TPW + t) * 16;

        // ----- A operand: 16 tokens x 64 dims, two K-steps of 32 -----
        const float* zrow = ze + (size_t)(tok0 + (lane & 15)) * DIM;
        v16bf a0 = load_a_slice(zrow, 0 + ka);    // dims  0..31
        v16bf a1 = load_a_slice(zrow, 32 + ka);   // dims 32..63

        // Packed (score | tag) running max per row. tag = 1023 - code so that
        // equal truncated scores resolve to the LOWER code under u32 max,
        // matching jnp.argmin's first-occurrence rule.
        unsigned bestp[8];
        #pragma unroll
        for (int r = 0; r < 8; ++r) bestp[r] = 0u;

        // ----- sweep 512 codes in 32 column tiles of 16 -----
        #pragma unroll 4
        for (int ct = 0; ct < 32; ++ct) {
            const int code = ct * 16 + col;
            const __bf16* erow = sE + code * DIM;
            v16bf b0 = *(const v16bf*)(erow + kb);        // dims kb..kb+15
            v16bf b1 = *(const v16bf*)(erow + 32 + kb);   // dims 32+kb..
            v8f c = {0.f, 0.f, 0.f, 0.f, 0.f, 0.f, 0.f, 0.f};
            c = __builtin_amdgcn_wmma_f32_16x16x32_bf16(false, a0, false, b0,
                                                        (short)0, c, false, false);
            c = __builtin_amdgcn_wmma_f32_16x16x32_bf16(false, a1, false, b1,
                                                        (short)0, c, false, false);
            const float nb = sN[code];              // (512 - ||e||^2)
            const unsigned tag = 1023u - (unsigned)code;
            #pragma unroll
            for (int r = 0; r < 8; ++r) {
                float s = fmaf(c[r], 2.0f, nb);     // 2*dot + 512 - ||e||^2 > 0
                unsigned u = (__float_as_uint(s) & 0xFFFFFC00u) | tag;
                bestp[r] = bestp[r] >= u ? bestp[r] : u;   // v_max_u32
            }
        }

        // ----- branch-free max reduction across each 16-lane half -----
        #pragma unroll
        for (int off = 1; off <= 8; off <<= 1) {
            #pragma unroll
            for (int r = 0; r < 8; ++r) {
                unsigned o = (unsigned)__shfl_xor((int)bestp[r], off, 32);
                bestp[r] = bestp[r] >= o ? bestp[r] : o;
            }
        }

        // ----- indices (rows 0-7 from lane 0, rows 8-15 from lane 16) -----
        if ((lane & 15) == 0) {
            const int rb = (lane & 16) ? 8 : 0;
            #pragma unroll
            for (int r = 0; r < 8; ++r)
                out_ix[tok0 + rb + r] = (float)(1023 - (int)(bestp[r] & 1023u));
        }

        // ----- gather zq (exact f32), store, accumulate loss -----
        // lanes 0-15 own rows 0-7 (2 lanes/row), lanes 16-31 own rows 8-15.
        const int mloc = ((lane & 16) ? 8 : 0) + (lane & 7);
        const int dimb = (lane & 8) ? 32 : 0;
        const int code = 1023 - (int)(bestp[lane & 7] & 1023u);
        const float4* e4 = (const float4*)(emb + code * DIM + dimb);
        const float4* z4 = (const float4*)(ze + (size_t)(tok0 + mloc) * DIM + dimb);
        float4* o4 = (float4*)(out + (size_t)(tok0 + mloc) * DIM + dimb);
        #pragma unroll
        for (int j = 0; j < 8; ++j) {
            float4 e = e4[j], z = z4[j];
            o4[j] = e;  // zq_st == zq numerically
            float dx = e.x - z.x, dy = e.y - z.y, dz = e.z - z.z, dw = e.w - z.w;
            lsum += dx * dx + dy * dy + dz * dz + dw * dw;
        }
    }

    // ----- wave loss reduction + one atomic per wave -----
    #pragma unroll
    for (int off = 16; off >= 1; off >>= 1)
        lsum += __shfl_xor(lsum, off, 32);
    if (lane == 0)
        atomicAdd(loss_acc, lsum);
}

__global__ void __launch_bounds__(256) vq_fin_kernel(float* __restrict__ out,
                                                     const float* __restrict__ loss_acc) {
    if (threadIdx.x == 0) {
        // vq_loss = mean((zq-ze)^2) * 2
        out[(size_t)N_TOK * DIM + N_TOK] =
            2.0f * (*loss_acc) / ((float)N_TOK * (float)DIM);
    }
}

extern "C" void kernel_launch(void* const* d_in, const int* in_sizes, int n_in,
                              void* d_out, int out_size, void* d_ws, size_t ws_size,
                              hipStream_t stream) {
    const float* ze  = (const float*)d_in[0];   // [524288, 64] f32
    const float* emb = (const float*)d_in[1];   // [512, 64]    f32
    float* out = (float*)d_out;                 // zq (N*64) | emb_ix (N) | loss (1)
    float* loss_acc = (float*)d_ws;

    vq_init_kernel<<<1, 256, 0, stream>>>(loss_acc);

    const int tiles  = N_TOK / 16;                    // 32768
    const int blocks = tiles / (WAVES * TPW);         // 1024
    const size_t shmem = (size_t)N_CODE * DIM * sizeof(__bf16) + N_CODE * sizeof(float);
    vq_main_kernel<<<blocks, 256, shmem, stream>>>(ze, emb, out, loss_acc);

    vq_fin_kernel<<<1, 256, 0, stream>>>(out, loss_acc);
}
